// SwinBlock_21251498180825
// MI455X (gfx1250) — matrix operand order repaired and checked by reading, not compiled, not run on hardware
//
#include <hip/hip_runtime.h>
#include <hip/hip_bf16.h>
#include <math.h>

// ---------------------------------------------------------------------------
// Types for CDNA5 WMMA (wave32): A/B = 16 bf16 per lane, C/D = 8 f32 per lane.
// ---------------------------------------------------------------------------
typedef __bf16 bf16;
typedef bf16  bf16x8  __attribute__((ext_vector_type(8)));
typedef bf16  bf16x16 __attribute__((ext_vector_type(16)));
typedef float f32x8   __attribute__((ext_vector_type(8)));
typedef float f32x4   __attribute__((ext_vector_type(4)));

#define DEV static __device__ __forceinline__

// Problem constants (Swin block, B=32, 56x56, C=256, 7x7 windows, shift 3)
constexpr int MROWS = 100352;     // 32 * 56 * 56 = 2048 windows * 49 tokens
constexpr int NWIN  = 2048;       // total windows
constexpr int NTOK  = 49;         // tokens per window
constexpr int NHEAD = 8;
constexpr int HDIM  = 32;

// ---------------------------------------------------------------------------
// WMMA fragment loaders, A/B in contiguous k order (same for both, not the ISA table).
// 16-bit A 16x32:  lane m=l&15, half h=l>>4; elem e -> K = e + 16h
// 16-bit B 32x16:  lane n=l&15, half h=l>>4; elem e -> K = e + 16h
// f32 C/D 16x16:   lane n=l&15; reg r -> M = r + 8*(l>>4)
// ---------------------------------------------------------------------------
DEV bf16x16 load_a_global(const bf16* p /*[row0,k0]*/, int ldk, int lane) {
  const int m = lane & 15, h = lane >> 4;
  const bf16* q = p + (size_t)m * ldk + h * 16;
  bf16x8 lo = *(const bf16x8*)q;          // K = 16h + 0..7
  bf16x8 hi = *(const bf16x8*)(q + 8);    // K = 16h + 8..15
  bf16x16 r;
#pragma unroll
  for (int e = 0; e < 8; ++e) { r[e] = lo[e]; r[e + 8] = hi[e]; }
  return r;
}

DEV bf16x16 load_b_global(const bf16* p /*[n0,k0] row-major, K contig*/, int ldk, int lane) {
  const int n = lane & 15, h = lane >> 4;
  return *(const bf16x16*)(p + (size_t)n * ldk + 16 * h);  // K = 16h + 0..15
}

// Guarded variants for the 49-row attention tiles (rows >= 49 read as zero so
// garbage can never feed NaNs into a WMMA).
DEV bf16x16 load49_a(const bf16* base /*[49,32]*/, int row0, int lane) {
  const int m = row0 + (lane & 15), h = lane >> 4;
  bf16x16 r;
  if (m < NTOK) {
    const bf16* p = base + m * HDIM + h * 16;
    bf16x8 lo = *(const bf16x8*)p;
    bf16x8 hi = *(const bf16x8*)(p + 8);
#pragma unroll
    for (int e = 0; e < 8; ++e) { r[e] = lo[e]; r[e + 8] = hi[e]; }
  } else {
#pragma unroll
    for (int e = 0; e < 16; ++e) r[e] = (bf16)0.f;
  }
  return r;
}

DEV bf16x16 load49_b(const bf16* base /*[49,32]*/, int n0, int lane) {
  const int n = n0 + (lane & 15), h = lane >> 4;
  bf16x16 r;
  if (n < NTOK) {
    r = *(const bf16x16*)(base + n * HDIM + 16 * h);
  } else {
#pragma unroll
    for (int e = 0; e < 16; ++e) r[e] = (bf16)0.f;
  }
  return r;
}

// shifted-window token t  ->  plain (b, y*56+x) row (same map for gather & scatter,
// since roll(+3) is the inverse of roll(-3)).
DEV size_t win_token_to_plain(int t) {
  const int win = t / NTOK, i = t - win * NTOK;
  const int b = win >> 6, w = win & 63;
  const int wy = w >> 3, wx = w & 7;
  const int ih = i / 7, iw = i - ih * 7;
  int y = wy * 7 + ih + 3; if (y >= 56) y -= 56;
  int x = wx * 7 + iw + 3; if (x >= 56) x -= 56;
  return (size_t)b * 3136 + y * 56 + x;
}

// region id for the shift-attention mask (rows/cols 0..48 | 49..52 | 53..55)
DEV int regf(int y) { return (y < 49) ? 0 : ((y < 53) ? 1 : 2); }

// ---------------------------------------------------------------------------
// fp32 -> bf16 weight convert
// ---------------------------------------------------------------------------
__global__ void cvt_f32_bf16(const float* __restrict__ s, bf16* __restrict__ d, int n) {
  int i = blockIdx.x * blockDim.x + threadIdx.x;
  if (i < n) d[i] = (bf16)s[i];
}

// ---------------------------------------------------------------------------
// LayerNorm: one wave per token (256 ch -> 8 per lane). GATHER=true fuses the
// cyclic shift + window partition (reads plain order, writes window order).
// ---------------------------------------------------------------------------
template <bool GATHER>
__global__ __launch_bounds__(256) void ln_kernel(const float* __restrict__ src,
                                                 const float* __restrict__ gamma,
                                                 const float* __restrict__ beta,
                                                 bf16* __restrict__ out) {
  const int t = blockIdx.x * 8 + (threadIdx.x >> 5);
  const int lane = threadIdx.x & 31;
  size_t srow = GATHER ? win_token_to_plain(t) : (size_t)t;

  const float* p = src + srow * 256 + lane * 8;
  f32x4 a = *(const f32x4*)p;
  f32x4 c = *(const f32x4*)(p + 4);
  float v[8] = {a[0], a[1], a[2], a[3], c[0], c[1], c[2], c[3]};

  float sum = 0.f;
#pragma unroll
  for (int e = 0; e < 8; ++e) sum += v[e];
#pragma unroll
  for (int m = 16; m >= 1; m >>= 1) sum += __shfl_xor(sum, m, 32);
  const float mean = sum * (1.f / 256.f);

  float sq = 0.f;
#pragma unroll
  for (int e = 0; e < 8; ++e) { float d = v[e] - mean; sq += d * d; }
#pragma unroll
  for (int m = 16; m >= 1; m >>= 1) sq += __shfl_xor(sq, m, 32);
  const float rs = rsqrtf(sq * (1.f / 256.f) + 1e-5f);

  bf16x8 o;
#pragma unroll
  for (int e = 0; e < 8; ++e) {
    int cix = lane * 8 + e;
    o[e] = (bf16)((v[e] - mean) * rs * gamma[cix] + beta[cix]);
  }
  *(bf16x8*)(out + (size_t)t * 256 + lane * 8) = o;
}

// ---------------------------------------------------------------------------
// Generic WMMA GEMM:  out[M, Nout] = A[M, K] * W[Nout, K]^T  (+ fused epilogue)
// 256 threads = 8 waves; workgroup tile 128x128; wave tile 32x64 (2x4 WMMAs).
// All tile dims divide exactly (M = 784*128, Nout in {256,768,1024}, K%32==0).
// A/B fragments are loaded straight from global: the 192MB L2 provides reuse.
// EPI: 0 = QKV scatter (+bias, q*scale, v transposed)
//      1 = proj (+bias, un-window/un-roll scatter, +shortcut) -> fp32 x2
//      2 = MLP1 (+bias, exact GELU) -> bf16 hidden
//      3 = MLP2 (+bias, +x2 residual) -> fp32 d_out
// ---------------------------------------------------------------------------
template <int KDIM, int EPI>
__global__ __launch_bounds__(256) void gemm_wmma(const bf16* __restrict__ A,
                                                 const bf16* __restrict__ Wt,
                                                 const float* __restrict__ bias,
                                                 bf16* __restrict__ ob0,
                                                 bf16* __restrict__ ob1,
                                                 bf16* __restrict__ ob2,
                                                 const float* __restrict__ aux,
                                                 float* __restrict__ outf) {
  const int lane = threadIdx.x & 31;
  const int wave = threadIdx.x >> 5;
  const int Nout = gridDim.x * 128;
  const int mof = blockIdx.y * 128 + (wave & 3) * 32;
  const int nof = blockIdx.x * 128 + (wave >> 2) * 64;

  f32x8 acc[2][4] = {};

  for (int k0 = 0; k0 < KDIM; k0 += 32) {
    bf16x16 af[2], bf[4];
    af[0] = load_a_global(A + (size_t)mof * KDIM + k0, KDIM, lane);
    af[1] = load_a_global(A + (size_t)(mof + 16) * KDIM + k0, KDIM, lane);
#pragma unroll
    for (int nt = 0; nt < 4; ++nt)
      bf[nt] = load_b_global(Wt + (size_t)(nof + nt * 16) * KDIM + k0, KDIM, lane);
#pragma unroll
    for (int mt = 0; mt < 2; ++mt)
#pragma unroll
      for (int nt = 0; nt < 4; ++nt)
        acc[mt][nt] = __builtin_amdgcn_wmma_f32_16x16x32_bf16(
            false, af[mt], false, bf[nt], (short)0, acc[mt][nt], false, false);
  }

  const int ln_ = lane & 15, hh = lane >> 4;
#pragma unroll
  for (int mt = 0; mt < 2; ++mt) {
#pragma unroll
    for (int nt = 0; nt < 4; ++nt) {
      const int col = nof + nt * 16 + ln_;
      const float bv = bias[col];
#pragma unroll
      for (int r = 0; r < 8; ++r) {
        const int row = mof + mt * 16 + r + 8 * hh;
        float v = acc[mt][nt][r] + bv;
        if (EPI == 0) {
          // QKV scatter: col = s*256 + h*32 + d ; q pre-scaled by hd^-0.5
          const int s = col >> 8, h = (col >> 5) & 7, d = col & 31;
          const int win = row / NTOK, ii = row - win * NTOK;
          const size_t id = (size_t)win * NHEAD + h;
          if (s == 0) {
            v *= 0.17677669529663687f;  // 1/sqrt(32)
            ob0[(id * NTOK + ii) * HDIM + d] = (bf16)v;
          } else if (s == 1) {
            ob1[(id * NTOK + ii) * HDIM + d] = (bf16)v;
          } else {
            ob2[(id * HDIM + d) * NTOK + ii] = (bf16)v;  // v stored transposed
          }
        } else if (EPI == 1) {
          const size_t prow = win_token_to_plain(row);
          outf[prow * 256 + col] = v + aux[prow * 256 + col];  // + shortcut x
        } else if (EPI == 2) {
          const float g = 0.5f * v * (1.f + erff(v * 0.70710678118654752f));
          ob0[(size_t)row * Nout + col] = (bf16)g;
        } else {
          outf[(size_t)row * 256 + col] = v + aux[(size_t)row * 256 + col];  // + x2
        }
      }
    }
  }
}

// ---------------------------------------------------------------------------
// Fused windowed attention: one wave per (window, head).
//   S = q k^T (16 WMMAs over 64x64 padded), + rel-pos bias gather + shift mask,
//   softmax with shfl reductions inside the 16-lane halves of the C layout,
//   P -> LDS (bf16, row stride 72 to de-conflict banks), reload as A fragments,
//   O = P v (16 WMMAs, K padded to 64, v columns >= 49 guarded to zero).
// ---------------------------------------------------------------------------
__global__ __launch_bounds__(128) void attn_kernel(const bf16* __restrict__ qb,
                                                   const bf16* __restrict__ kb,
                                                   const bf16* __restrict__ vt,
                                                   const float* __restrict__ bias_table,
                                                   bf16* __restrict__ attnb) {
  __shared__ bf16 plds[4][64 * 72];
  const int lane = threadIdx.x & 31;
  const int wv = threadIdx.x >> 5;
  const int id = blockIdx.x * 4 + wv;     // win*8 + head
  const int win = id >> 3, head = id & 7;
  const int wy = (win & 63) >> 3, wx = win & 7;
  const int ln_ = lane & 15, hh = lane >> 4;

  const bf16* qp = qb + (size_t)id * NTOK * HDIM;
  const bf16* kp = kb + (size_t)id * NTOK * HDIM;
  const bf16* vp = vt + (size_t)id * HDIM * NTOK;  // [d][i]

  // ---- S = q k^T ----
  bf16x16 aq[4], bk[4];
#pragma unroll
  for (int mt = 0; mt < 4; ++mt) aq[mt] = load49_a(qp, mt * 16, lane);
#pragma unroll
  for (int nt = 0; nt < 4; ++nt) bk[nt] = load49_b(kp, nt * 16, lane);
  f32x8 S[4][4] = {};
#pragma unroll
  for (int mt = 0; mt < 4; ++mt)
#pragma unroll
    for (int nt = 0; nt < 4; ++nt)
      S[mt][nt] = __builtin_amdgcn_wmma_f32_16x16x32_bf16(
          false, aq[mt], false, bk[nt], (short)0, S[mt][nt], false, false);

  // ---- column (key) info per lane ----
  int jok[4], jh[4], jw[4], jreg[4];
#pragma unroll
  for (int nt = 0; nt < 4; ++nt) {
    const int j = nt * 16 + ln_;
    jok[nt] = (j < NTOK);
    jh[nt] = j / 7; jw[nt] = j - jh[nt] * 7;
    jreg[nt] = regf(wy * 7 + jh[nt]) * 3 + regf(wx * 7 + jw[nt]);
  }

  // ---- bias + mask + softmax, write P to LDS ----
  bf16* P = plds[wv];
#pragma unroll
  for (int mt = 0; mt < 4; ++mt) {
#pragma unroll
    for (int r = 0; r < 8; ++r) {
      const int i = mt * 16 + r + 8 * hh;
      const bool iv = (i < NTOK);
      const int ih = i / 7, iw = i - ih * 7;
      const int ireg = regf(wy * 7 + ih) * 3 + regf(wx * 7 + iw);
      float sv[4];
      float mx = -3e38f;
#pragma unroll
      for (int nt = 0; nt < 4; ++nt) {
        float s = -1e30f;
        if (iv && jok[nt]) {
          const int rel = (ih - jh[nt] + 6) * 13 + (iw - jw[nt] + 6);
          const float bsv = bias_table[rel * NHEAD + head];
          const float msk = (ireg != jreg[nt]) ? -100.f : 0.f;
          s = S[mt][nt][r] + bsv + msk;
        }
        sv[nt] = s;
        mx = fmaxf(mx, s);
      }
      // row reductions stay inside the 16-lane half holding this row
      mx = fmaxf(mx, __shfl_xor(mx, 1, 32));
      mx = fmaxf(mx, __shfl_xor(mx, 2, 32));
      mx = fmaxf(mx, __shfl_xor(mx, 4, 32));
      mx = fmaxf(mx, __shfl_xor(mx, 8, 32));
      float sum = 0.f;
#pragma unroll
      for (int nt = 0; nt < 4; ++nt) { sv[nt] = __expf(sv[nt] - mx); sum += sv[nt]; }
      sum += __shfl_xor(sum, 1, 32);
      sum += __shfl_xor(sum, 2, 32);
      sum += __shfl_xor(sum, 4, 32);
      sum += __shfl_xor(sum, 8, 32);
      const float inv = 1.f / sum;
#pragma unroll
      for (int nt = 0; nt < 4; ++nt)
        P[i * 72 + nt * 16 + ln_] = (bf16)(sv[nt] * inv);
    }
  }
  asm volatile("s_wait_dscnt 0" ::: "memory");

  // ---- O = P v ----
  f32x8 O[4][2] = {};
#pragma unroll
  for (int ks = 0; ks < 2; ++ks) {
    bf16x16 ap[4];
#pragma unroll
    for (int mt = 0; mt < 4; ++mt) {
      const bf16* pp = P + (mt * 16 + ln_) * 72 + ks * 32 + hh * 16;
      bf16x8 lo = *(const bf16x8*)pp;
      bf16x8 hi = *(const bf16x8*)(pp + 8);
#pragma unroll
      for (int e = 0; e < 8; ++e) { ap[mt][e] = lo[e]; ap[mt][e + 8] = hi[e]; }
    }
    bf16x16 bv[2];
#pragma unroll
    for (int nt = 0; nt < 2; ++nt) {
      const int d = nt * 16 + ln_;
      const bf16* pp = vp + d * NTOK;
#pragma unroll
      for (int e = 0; e < 16; ++e) {
        const int kk = ks * 32 + 16 * hh + e;
        bv[nt][e] = (kk < NTOK) ? pp[kk] : (bf16)0.f;
      }
    }
#pragma unroll
    for (int mt = 0; mt < 4; ++mt)
#pragma unroll
      for (int nt = 0; nt < 2; ++nt)
        O[mt][nt] = __builtin_amdgcn_wmma_f32_16x16x32_bf16(
            false, ap[mt], false, bv[nt], (short)0, O[mt][nt], false, false);
  }

  // ---- store to window-ordered [t, C] bf16 (input of proj GEMM) ----
#pragma unroll
  for (int mt = 0; mt < 4; ++mt)
#pragma unroll
    for (int nt = 0; nt < 2; ++nt)
#pragma unroll
      for (int r = 0; r < 8; ++r) {
        const int i = mt * 16 + r + 8 * hh;
        if (i < NTOK) {
          const int d = nt * 16 + ln_;
          attnb[((size_t)win * NTOK + i) * 256 + head * HDIM + d] = (bf16)O[mt][nt][r];
        }
      }
}

// ---------------------------------------------------------------------------
// Host-side launch sequence
// ---------------------------------------------------------------------------
extern "C" void kernel_launch(void* const* d_in, const int* in_sizes, int n_in,
                              void* d_out, int out_size, void* d_ws, size_t ws_size,
                              hipStream_t stream) {
  (void)in_sizes; (void)n_in; (void)out_size;
  const float* x          = (const float*)d_in[0];
  const float* g1         = (const float*)d_in[1];
  const float* b1         = (const float*)d_in[2];
  const float* qkv_w      = (const float*)d_in[3];
  const float* qkv_b      = (const float*)d_in[4];
  const float* bias_table = (const float*)d_in[5];
  const float* proj_w     = (const float*)d_in[6];
  const float* proj_b     = (const float*)d_in[7];
  const float* g2         = (const float*)d_in[8];
  const float* b2         = (const float*)d_in[9];
  const float* w1         = (const float*)d_in[10];
  const float* bm1        = (const float*)d_in[11];
  const float* w2         = (const float*)d_in[12];
  const float* bm2        = (const float*)d_in[13];

  // workspace layout (bytes): [bf16 weights | aln | q | k | vT | attn | x2]
  // hidden (M*1024 bf16) exactly overlays the dead q/k/vT/attn region.
  const size_t M = MROWS;
  char* ws = (char*)d_ws;
  bf16* wqkv  = (bf16*)ws;                    // 768*256
  bf16* wproj = wqkv + 768 * 256;             // 256*256
  bf16* w1b   = wproj + 256 * 256;            // 1024*256
  bf16* w2b   = w1b + 1024 * 256;             // 256*1024
  bf16* aln   = (bf16*)(ws + (2u << 20));     // M*256 bf16
  bf16* qb    = aln + M * 256;
  bf16* kb    = qb + M * 256;
  bf16* vb    = kb + M * 256;
  bf16* attnb = vb + M * 256;
  bf16* hid   = qb;                           // overlay: M*1024 bf16
  float* x2   = (float*)(attnb + M * 256);    // M*256 fp32

  const size_t needed = (2u << 20) + M * 256 * 2 + M * 1024 * 2 + M * 256 * 4;
  if (ws_size < needed) return;

  // 1) weights -> bf16
  cvt_f32_bf16<<<(768 * 256 + 255) / 256, 256, 0, stream>>>(qkv_w, wqkv, 768 * 256);
  cvt_f32_bf16<<<(256 * 256 + 255) / 256, 256, 0, stream>>>(proj_w, wproj, 256 * 256);
  cvt_f32_bf16<<<(1024 * 256 + 255) / 256, 256, 0, stream>>>(w1, w1b, 1024 * 256);
  cvt_f32_bf16<<<(1024 * 256 + 255) / 256, 256, 0, stream>>>(w2, w2b, 1024 * 256);

  // 2) LN1 fused with shift + window partition
  ln_kernel<true><<<MROWS / 8, 256, 0, stream>>>(x, g1, b1, aln);

  // 3) QKV GEMM (M x 256 x 768) + q-scale + q/k/vT scatter
  gemm_wmma<256, 0><<<dim3(6, 784), 256, 0, stream>>>(
      aln, wqkv, qkv_b, qb, kb, vb, nullptr, nullptr);

  // 4) fused windowed attention (2048 windows * 8 heads, 4 per block)
  attn_kernel<<<4096, 128, 0, stream>>>(qb, kb, vb, bias_table, attnb);

  // 5) proj GEMM (M x 256 x 256) + un-window/un-roll + shortcut -> x2 (fp32)
  gemm_wmma<256, 1><<<dim3(2, 784), 256, 0, stream>>>(
      attnb, wproj, proj_b, nullptr, nullptr, nullptr, x, x2);

  // 6) LN2
  ln_kernel<false><<<MROWS / 8, 256, 0, stream>>>(x2, g2, b2, aln);

  // 7) MLP1 (M x 256 x 1024) + exact GELU
  gemm_wmma<256, 2><<<dim3(8, 784), 256, 0, stream>>>(
      aln, w1b, bm1, hid, nullptr, nullptr, nullptr, nullptr);

  // 8) MLP2 (M x 1024 x 256) + x2 residual -> d_out
  gemm_wmma<1024, 3><<<dim3(2, 784), 256, 0, stream>>>(
      hid, w2b, bm2, nullptr, nullptr, nullptr, x2, (float*)d_out);
}
